// Encoder_GRU_2_33062658245302
// MI455X (gfx1250) — compile-verified, matrix operand order checked by reading, not run on hardware
//
#include <hip/hip_runtime.h>
#include <hip/hip_bf16.h>
#include <stdint.h>

typedef __attribute__((ext_vector_type(16))) _Float16 v16h;
typedef __attribute__((ext_vector_type(8)))  _Float16 v8h;
typedef __attribute__((ext_vector_type(4)))  _Float16 v4h;
typedef __attribute__((ext_vector_type(8)))  float    v8f;
typedef int i128 __attribute__((vector_size(16)));   // matches async builtin param

#define B_SZ   256
#define T_SZ   128
#define EMB_SZ 1024
#define HID_SZ 1024
#define G_SZ   3072
#define NTOK   (B_SZ * T_SZ)   // 32768

#define AS1 __attribute__((address_space(1)))
#define AS3 __attribute__((address_space(3)))

#if __has_builtin(__builtin_amdgcn_global_load_async_to_lds_b128) && \
    __has_builtin(__builtin_amdgcn_s_wait_asynccnt)
#define USE_ASYNC 1
#else
#define USE_ASYNC 0
#endif

// ---------------------------------------------------------------------------
// WMMA fragment helpers (layouts per CDNA5 ISA 7.12.2, 16-bit A 16x32)
// lane<16 : row M=lane,    K elements {k0+0..7, k0+16..23}
// lane>=16: row M=lane-16, K elements {k0+8..15, k0+24..31}
// ---------------------------------------------------------------------------
__device__ __forceinline__ int kidx(int lane, int e) {
  return ((lane & 16) ? 8 : 0) + ((e < 8) ? e : (e + 8));
}

__device__ __forceinline__ v16h make_a(v8h a0, v8h a1) {
  v16h a;
#pragma unroll
  for (int i = 0; i < 8; ++i) { a[i] = a0[i]; a[i + 8] = a1[i]; }
  return a;
}

__device__ __forceinline__ v16h load_a_frag(const _Float16* __restrict__ rowk0, int lane) {
  const int lo = (lane & 16) ? 8 : 0;
  v8h a0 = *reinterpret_cast<const v8h*>(rowk0 + lo);
  v8h a1 = *reinterpret_cast<const v8h*>(rowk0 + lo + 16);
  return make_a(a0, a1);
}

__device__ __forceinline__ v8f wmma_f16(v16h a, v16h b, v8f c) {
  return __builtin_amdgcn_wmma_f32_16x16x32_f16(false, a, false, b, (short)0, c,
                                                false, false);
}

// ---------------------------------------------------------------------------
__global__ __launch_bounds__(256) void zero_kernel(uint32_t* __restrict__ p, int n) {
  int i = blockIdx.x * blockDim.x + threadIdx.x;
  if (i < n) p[i] = 0u;
}

// ---------------------------------------------------------------------------
// Embedding gather: x16[tok,:] = (f16) emb[id,:]; mask[tok] = id != 0
// ---------------------------------------------------------------------------
__global__ __launch_bounds__(256) void embed_kernel(const float* __restrict__ emb,
                                                    const int* __restrict__ ids,
                                                    _Float16* __restrict__ x16,
                                                    float* __restrict__ maskOut) {
  int tok = blockIdx.x;
  int id  = ids[tok];
  if (threadIdx.x == 0) maskOut[tok] = (id != 0) ? 1.0f : 0.0f;
  const float4* src = reinterpret_cast<const float4*>(emb + (size_t)id * EMB_SZ);
  float4 v = src[threadIdx.x];
  v4h h; h[0] = (_Float16)v.x; h[1] = (_Float16)v.y; h[2] = (_Float16)v.z; h[3] = (_Float16)v.w;
  *reinterpret_cast<v4h*>(x16 + (size_t)tok * EMB_SZ + threadIdx.x * 4) = h;
}

// ---------------------------------------------------------------------------
// Pack row-major f32 weight [K,N] into f16 WMMA-B tile layout.
// ---------------------------------------------------------------------------
__global__ __launch_bounds__(256) void pack_b_kernel(const float* __restrict__ W,
                                                     _Float16* __restrict__ Wp,
                                                     int K, int N) {
  int wid  = (blockIdx.x * blockDim.x + threadIdx.x) >> 5;
  int lane = threadIdx.x & 31;
  int K32  = K >> 5;
  int nblk = (N >> 4) * K32;
  if (wid >= nblk) return;
  int ntile = wid / K32;
  int kc    = wid % K32;
  int col   = ntile * 16 + (lane & 15);
  int k0    = kc * 32;
  _Float16* dst = Wp + (size_t)wid * 512 + lane * 16;
#pragma unroll
  for (int e = 0; e < 16; ++e) {
    int k = k0 + kidx(lane, e);
    dst[e] = (_Float16)W[(size_t)k * N + col];
  }
}

// ---------------------------------------------------------------------------
// GEMM: C[M,N](f32) = A[M,K](f16 row-major) @ Bp(packed) + bias[N]
// Wave tile 64x32: 8 accumulators; A chunk staged in LDS via async copy,
// shared by all 8 waves of the block (same 64 rows, different n-pairs).
// Branch-free steady-state pipeline (depth 2) + explicit 2-iter epilogue.
// ---------------------------------------------------------------------------
#define AROWH 40           // padded LDS row stride in halves (80 B): conflict-free
#define ABUFH (64 * AROWH) // halves per chunk buffer

__global__ __launch_bounds__(256) void gemm_kernel(const _Float16* __restrict__ A,
                                                   const _Float16* __restrict__ Bp,
                                                   const float* __restrict__ bias,
                                                   float* __restrict__ C,
                                                   int M, int N, int K) {
  const int lane = threadIdx.x & 31;
  const int w    = threadIdx.x >> 5;
  const int K32    = K >> 5;
  const int npairs = N >> 5;
  int wid   = blockIdx.x * 8 + w;
  int np    = wid % npairs;
  int mquad = wid / npairs;      // uniform across the block (npairs % 8 == 0)
  int m0    = mquad * 64;

  v8f acc[8];
#pragma unroll
  for (int i = 0; i < 8; ++i) acc[i] = (v8f){};

  const _Float16* b0 = Bp + (size_t)(np * 2    ) * K32 * 512 + lane * 16;
  const _Float16* b1 = Bp + (size_t)(np * 2 + 1) * K32 * 512 + lane * 16;
  const int lo = (lane & 16) ? 8 : 0;

#if USE_ASYNC
  __shared__ _Float16 sA[4 * ABUFH];
  const int crow = w * 8 + (lane >> 2);       // row copied by this lane
  const int cseg = (lane & 3) * 8;            // 16-byte segment within row
  const _Float16* gsrc0 = A + (size_t)(m0 + crow) * K + cseg;
  _Float16* ldst0 = &sA[crow * AROWH + cseg];
  auto issueA = [&](int kc) {
    __builtin_amdgcn_global_load_async_to_lds_b128(
        (AS1 i128*)(gsrc0 + (size_t)kc * 32),
        (AS3 i128*)(ldst0 + (kc & 3) * ABUFH), 0, 0);
  };
#endif

  auto compute = [&](int kc) {
    v16h vb0 = *reinterpret_cast<const v16h*>(b0 + (size_t)kc * 512);
    v16h vb1 = *reinterpret_cast<const v16h*>(b1 + (size_t)kc * 512);
#pragma unroll
    for (int i = 0; i < 4; ++i) {
#if USE_ASYNC
      const _Float16* srow = &sA[(kc & 3) * ABUFH + (i * 16 + (lane & 15)) * AROWH + lo];
      v8h x0 = *reinterpret_cast<const v8h*>(srow);
      v8h x1 = *reinterpret_cast<const v8h*>(srow + 16);
      v16h a = make_a(x0, x1);
#else
      v16h a = load_a_frag(A + (size_t)(m0 + i * 16 + (lane & 15)) * K + kc * 32, lane);
#endif
      acc[i * 2    ] = wmma_f16(a, vb0, acc[i * 2    ]);
      acc[i * 2 + 1] = wmma_f16(a, vb1, acc[i * 2 + 1]);
    }
  };

#if USE_ASYNC
  issueA(0);
  issueA(1);
  int kc = 0;
  for (; kc + 2 < K32; ++kc) {       // branch-free steady state
    issueA(kc + 2);
    __builtin_amdgcn_s_wait_asynccnt(2);
    __syncthreads();
    compute(kc);
  }
  __builtin_amdgcn_s_wait_asynccnt(1);   // kc == K32-2
  __syncthreads();
  compute(K32 - 2);
  __builtin_amdgcn_s_wait_asynccnt(0);   // kc == K32-1
  __syncthreads();
  compute(K32 - 1);
#else
  for (int kc = 0; kc < K32; ++kc) compute(kc);
#endif

  int rbase = m0 + ((lane >> 4) << 3);
#pragma unroll
  for (int j = 0; j < 2; ++j) {
    int col  = np * 32 + j * 16 + (lane & 15);
    float bv = bias[col];
#pragma unroll
    for (int i = 0; i < 4; ++i) {
#pragma unroll
      for (int v = 0; v < 8; ++v) {
        C[(size_t)(rbase + i * 16 + v) * N + col] = acc[i * 2 + j][v] + bv;
      }
    }
  }
}

// ---------------------------------------------------------------------------
// Fused GRU step: hg = h @ Wh (WMMA) ; gates ; masked state carry.
// Waves in a block share one hidden-column slice -> identical weight stream,
// async-staged into LDS (3 gate fragments / chunk, 4-buffer ring).
// waves = (B/16)*(HID/16) = 1024; h16 double-buffered (cross-wave RAW).
// ---------------------------------------------------------------------------
__global__ __launch_bounds__(256) void gru_step_kernel(
    const float* __restrict__ xg,            // [NTOK, G] input gates (bias applied)
    const _Float16* __restrict__ Whp,        // packed recurrent weights
    const float* __restrict__ bh,            // recurrent bias, length G
    const int* __restrict__ ids,             // [B,T] token ids (mask source)
    float* __restrict__ h32,                 // [B,HID] hidden state (f32)
    float* __restrict__ o32,                 // [B,HID] carried output state
    const _Float16* __restrict__ h16_in,     // [B,HID] prev hidden (f16)
    _Float16* __restrict__ h16_out,          // [B,HID] next hidden (f16)
    int tt,                                  // actual time index
    float* __restrict__ seqOut, int seqColOff,
    _Float16* __restrict__ seq16, int seq16Stride, int seq16ColOff) {
  const int lane = threadIdx.x & 31;
  const int NSL = HID_SZ / 16;   // 64 col slices
  const int K32 = HID_SZ / 32;   // 32 k-chunks
  int wid   = blockIdx.x * 8 + (threadIdx.x >> 5);
  int mtile = wid & 15;          // fastest: block shares nsl (weight stream)
  int nsl   = wid >> 4;
  int m0    = mtile * 16;

  v8f az = {}, ar = {}, ah = {};
  const _Float16* arow = h16_in + (size_t)(m0 + (lane & 15)) * HID_SZ;

#if USE_ASYNC
  __shared__ _Float16 sW[4 * 3 * 512];
  const int tid  = threadIdx.x;
  const bool cp  = tid < 192;                // waves 0..5 copy (no divergence)
  const int gate = (tid >> 6) & 3;
  const int idx  = tid & 63;                 // 16-byte segment
  const _Float16* wsrc0 = Whp + ((size_t)(nsl + gate * NSL) * K32) * 512 + idx * 8;
  _Float16* wdst0 = &sW[gate * 512 + idx * 8];
  auto issueW = [&](int kc) {
    if (cp)
      __builtin_amdgcn_global_load_async_to_lds_b128(
          (AS1 i128*)(wsrc0 + (size_t)kc * 512),
          (AS3 i128*)(wdst0 + (kc & 3) * 1536), 0, 0);
  };
#else
  const _Float16* bzp = Whp + (size_t)(nsl          ) * K32 * 512 + lane * 16;
  const _Float16* brp = Whp + (size_t)(nsl +     NSL) * K32 * 512 + lane * 16;
  const _Float16* bnp = Whp + (size_t)(nsl + 2 * NSL) * K32 * 512 + lane * 16;
#endif

  auto compute = [&](int kc) {
#if USE_ASYNC
    const _Float16* sb = &sW[(kc & 3) * 1536];
    v16h vz = *reinterpret_cast<const v16h*>(sb + lane * 16);
    v16h vr = *reinterpret_cast<const v16h*>(sb + 512 + lane * 16);
    v16h vh = *reinterpret_cast<const v16h*>(sb + 1024 + lane * 16);
#else
    v16h vz = *reinterpret_cast<const v16h*>(bzp + (size_t)kc * 512);
    v16h vr = *reinterpret_cast<const v16h*>(brp + (size_t)kc * 512);
    v16h vh = *reinterpret_cast<const v16h*>(bnp + (size_t)kc * 512);
#endif
    v16h a = load_a_frag(arow + kc * 32, lane);
    az = wmma_f16(a, vz, az);
    ar = wmma_f16(a, vr, ar);
    ah = wmma_f16(a, vh, ah);
  };

#if USE_ASYNC
  issueW(0);
  issueW(1);
  int kc = 0;
  for (; kc + 2 < K32; ++kc) {       // branch-free steady state
    issueW(kc + 2);
    __builtin_amdgcn_s_wait_asynccnt(2);
    __syncthreads();
    compute(kc);
  }
  __builtin_amdgcn_s_wait_asynccnt(1);
  __syncthreads();
  compute(K32 - 2);
  __builtin_amdgcn_s_wait_asynccnt(0);
  __syncthreads();
  compute(K32 - 1);
#else
  for (int kc = 0; kc < K32; ++kc) compute(kc);
#endif

  int col   = nsl * 16 + (lane & 15);
  float bbz = bh[col];
  float bbr = bh[HID_SZ + col];
  float bbh = bh[2 * HID_SZ + col];
#pragma unroll
  for (int v = 0; v < 8; ++v) {
    int b = m0 + ((lane >> 4) << 3) + v;               // batch row
    size_t tok = (size_t)b * T_SZ + tt;
    const float* xrow = xg + tok * G_SZ;
    float z  = 1.0f / (1.0f + __expf(-(xrow[col] + az[v] + bbz)));
    float r  = 1.0f / (1.0f + __expf(-(xrow[HID_SZ + col] + ar[v] + bbr)));
    float hh = tanhf(xrow[2 * HID_SZ + col] + r * (ah[v] + bbh));
    size_t hidx = (size_t)b * HID_SZ + col;
    float hold = h32[hidx];
    float hn   = z * hold + (1.0f - z) * hh;
    bool  mk   = (ids[tok] != 0);
    float hnew = mk ? hn : hold;
    float oold = o32[hidx];
    float onew = mk ? hn : oold;
    h32[hidx] = hnew;
    o32[hidx] = onew;
    h16_out[hidx] = (_Float16)hnew;
    seqOut[tok * (size_t)G_SZ + seqColOff + col] = onew;
    if (seq16) seq16[tok * (size_t)seq16Stride + seq16ColOff + col] = (_Float16)onew;
  }
}

// ---------------------------------------------------------------------------
// enc_hid = concat(h_f, h_b, h2)  [B, 3H]
// ---------------------------------------------------------------------------
__global__ __launch_bounds__(256) void enc_kernel(const float* __restrict__ hf,
                                                  const float* __restrict__ hb,
                                                  const float* __restrict__ h2,
                                                  float* __restrict__ enc) {
  int i = blockIdx.x * blockDim.x + threadIdx.x;     // 0 .. B*G-1
  int b = i / G_SZ;
  int j = i % G_SZ;
  float v;
  if (j < HID_SZ)            v = hf[(size_t)b * HID_SZ + j];
  else if (j < 2 * HID_SZ)   v = hb[(size_t)b * HID_SZ + j - HID_SZ];
  else                       v = h2[(size_t)b * HID_SZ + j - 2 * HID_SZ];
  enc[i] = v;
}

// ---------------------------------------------------------------------------
extern "C" void kernel_launch(void* const* d_in, const int* in_sizes, int n_in,
                              void* d_out, int out_size, void* d_ws, size_t ws_size,
                              hipStream_t stream) {
  (void)in_sizes; (void)n_in; (void)out_size; (void)ws_size;
  const float* emb  = (const float*)d_in[0];
  const float* Wi_f = (const float*)d_in[1];
  const float* Wh_f = (const float*)d_in[2];
  const float* b_f  = (const float*)d_in[3];
  const float* Wi_b = (const float*)d_in[4];
  const float* Wh_b = (const float*)d_in[5];
  const float* b_b  = (const float*)d_in[6];
  const float* Wi2  = (const float*)d_in[7];
  const float* Wh2  = (const float*)d_in[8];
  const float* b2   = (const float*)d_in[9];
  const int*   ids  = (const int*)d_in[10];
  float* out = (float*)d_out;

  // output layout: sequences [B,T,3H] | enc_hid [B,3H] | mask [B,T]
  float* seqOut  = out;
  float* encOut  = out + (size_t)NTOK * G_SZ;
  float* maskOut = encOut + (size_t)B_SZ * G_SZ;

  // workspace carve-out
  char* ws = (char*)d_ws;
  size_t off = 0;
  auto carve = [&](size_t bytes) -> char* {
    char* p = ws + off;
    off += (bytes + 255) & ~(size_t)255;
    return p;
  };
  _Float16* x16    = (_Float16*)carve((size_t)NTOK * EMB_SZ * 2);
  _Float16* seq116 = (_Float16*)carve((size_t)NTOK * 2 * HID_SZ * 2);
  float*    xg     = (float*)   carve((size_t)NTOK * G_SZ * 4);
  _Float16* wiFp   = (_Float16*)carve((size_t)EMB_SZ * G_SZ * 2);
  _Float16* whFp   = (_Float16*)carve((size_t)HID_SZ * G_SZ * 2);
  _Float16* wiBp   = (_Float16*)carve((size_t)EMB_SZ * G_SZ * 2);
  _Float16* whBp   = (_Float16*)carve((size_t)HID_SZ * G_SZ * 2);
  _Float16* wi2p   = (_Float16*)carve((size_t)2 * HID_SZ * G_SZ * 2);
  _Float16* wh2p   = (_Float16*)carve((size_t)HID_SZ * G_SZ * 2);
  float* hF = (float*)carve((size_t)B_SZ * HID_SZ * 4);
  float* oF = (float*)carve((size_t)B_SZ * HID_SZ * 4);
  float* hB = (float*)carve((size_t)B_SZ * HID_SZ * 4);
  float* oB = (float*)carve((size_t)B_SZ * HID_SZ * 4);
  float* h2 = (float*)carve((size_t)B_SZ * HID_SZ * 4);
  float* o2 = (float*)carve((size_t)B_SZ * HID_SZ * 4);
  _Float16* h16F[2] = { (_Float16*)carve((size_t)B_SZ * HID_SZ * 2),
                        (_Float16*)carve((size_t)B_SZ * HID_SZ * 2) };
  _Float16* h16B[2] = { (_Float16*)carve((size_t)B_SZ * HID_SZ * 2),
                        (_Float16*)carve((size_t)B_SZ * HID_SZ * 2) };
  _Float16* h162[2] = { (_Float16*)carve((size_t)B_SZ * HID_SZ * 2),
                        (_Float16*)carve((size_t)B_SZ * HID_SZ * 2) };

  // ---- weight packing (f32 -> f16 WMMA-B layout) ----
  auto pack = [&](const float* W, _Float16* Wp, int K, int N) {
    int nblk = (N >> 4) * (K >> 5);
    pack_b_kernel<<<nblk / 8, 256, 0, stream>>>(W, Wp, K, N);
  };
  pack(Wi_f, wiFp, EMB_SZ, G_SZ);
  pack(Wh_f, whFp, HID_SZ, G_SZ);
  pack(Wi_b, wiBp, EMB_SZ, G_SZ);
  pack(Wh_b, whBp, HID_SZ, G_SZ);
  pack(Wi2,  wi2p, 2 * HID_SZ, G_SZ);
  pack(Wh2,  wh2p, HID_SZ, G_SZ);

  // ---- zero state buffers ----
  auto zf32 = [&](float* p)    { zero_kernel<<<(B_SZ * HID_SZ) / 256, 256, 0, stream>>>((uint32_t*)p, B_SZ * HID_SZ); };
  auto zf16 = [&](_Float16* p) { zero_kernel<<<(B_SZ * HID_SZ / 2) / 256, 256, 0, stream>>>((uint32_t*)p, B_SZ * HID_SZ / 2); };
  zf32(hF); zf32(oF); zf32(hB); zf32(oB); zf32(h2); zf32(o2);
  zf16(h16F[0]); zf16(h16F[1]); zf16(h16B[0]); zf16(h16B[1]); zf16(h162[0]); zf16(h162[1]);

  // ---- embedding gather + mask ----
  embed_kernel<<<NTOK, 256, 0, stream>>>(emb, ids, x16, maskOut);

  const int gemmBlocks = (NTOK / 64) * (G_SZ / 32) / 8;   // 6144
  const int stepBlocks = (B_SZ / 16) * (HID_SZ / 16) / 8; // 128

  // ---- layer 1 forward ----
  gemm_kernel<<<gemmBlocks, 256, 0, stream>>>(x16, wiFp, b_f, xg, NTOK, G_SZ, EMB_SZ);
  for (int s = 0; s < T_SZ; ++s) {
    gru_step_kernel<<<stepBlocks, 256, 0, stream>>>(
        xg, whFp, b_f + G_SZ, ids, hF, oF, h16F[s & 1], h16F[(s + 1) & 1],
        /*tt=*/s, seqOut, /*seqColOff=*/0, seq116, 2 * HID_SZ, /*seq16off=*/0);
  }

  // ---- layer 1 backward ----
  gemm_kernel<<<gemmBlocks, 256, 0, stream>>>(x16, wiBp, b_b, xg, NTOK, G_SZ, EMB_SZ);
  for (int s = 0; s < T_SZ; ++s) {
    int tt = T_SZ - 1 - s;
    gru_step_kernel<<<stepBlocks, 256, 0, stream>>>(
        xg, whBp, b_b + G_SZ, ids, hB, oB, h16B[s & 1], h16B[(s + 1) & 1],
        tt, seqOut, /*seqColOff=*/HID_SZ, seq116, 2 * HID_SZ, /*seq16off=*/HID_SZ);
  }

  // ---- layer 2 (forward over concat(seq_f, seq_b)) ----
  gemm_kernel<<<gemmBlocks, 256, 0, stream>>>(seq116, wi2p, b2, xg, NTOK, G_SZ, 2 * HID_SZ);
  for (int s = 0; s < T_SZ; ++s) {
    gru_step_kernel<<<stepBlocks, 256, 0, stream>>>(
        xg, wh2p, b2 + G_SZ, ids, h2, o2, h162[s & 1], h162[(s + 1) & 1],
        /*tt=*/s, seqOut, /*seqColOff=*/2 * HID_SZ, (_Float16*)nullptr, 0, 0);
  }

  // ---- final states ----
  enc_kernel<<<(B_SZ * G_SZ) / 256, 256, 0, stream>>>(hF, hB, h2, encOut);
}
